// GRUModel_30365418783002
// MI455X (gfx1250) — matrix operand order repaired and checked by reading, not compiled, not run on hardware
//
#include <hip/hip_runtime.h>

#define GAS __attribute__((address_space(1)))
#define LAS __attribute__((address_space(3)))

typedef _Float16 v16h __attribute__((ext_vector_type(16)));
typedef _Float16 v8h  __attribute__((ext_vector_type(8)));
typedef _Float16 h4   __attribute__((ext_vector_type(4)));
typedef float    v8f  __attribute__((ext_vector_type(8)));
typedef float    f4   __attribute__((ext_vector_type(4)));
typedef int      i4v  __attribute__((ext_vector_type(4)));

constexpr int B  = 128;
constexpr int T  = 512;
constexpr int H  = 512;
constexpr int NG = 3 * H;            // 1536 gate columns
constexpr int WPER = NG * H;         // 786432 elements per weight matrix
constexpr int GSTR = 1036;           // f32 row stride, r/z sum buffer
constexpr int NSTR = 520;            // f16 row stride, i_n / h_n buffers
constexpr int HSTR = 520;            // f16 row stride, hidden-state buffers
constexpr int XSTR = 516;            // f32 row stride, staged x_t tile

// Async global->LDS prefetch path (CDNA5, ASYNCcnt). Compile-time probed so the
// file builds on host pass / toolchains without the builtin.
#if defined(__gfx1250__) && __has_builtin(__builtin_amdgcn_global_load_async_to_lds_b128) && __has_builtin(__builtin_amdgcn_s_wait_asynccnt)
#define HAVE_ASYNC_LDS 1
#else
#define HAVE_ASYNC_LDS 0
#endif

static __device__ __forceinline__ v8f wmma16x16x32(v16h a, v16h b, v8f c) {
  // D = A(16x32 f16) * B(32x16 f16) + C(16x16 f32)
  return __builtin_amdgcn_wmma_f32_16x16x32_f16(false, a, false, b, (short)0, c, false, false);
}

static __device__ __forceinline__ v16h cat16(v8h lo, v8h hi) {
  return __builtin_shufflevector(lo, hi, 0,1,2,3,4,5,6,7,8,9,10,11,12,13,14,15);
}

// A fragment (16x32 f16) from LDS f16 matrix, row stride HSTR.
// Lane m = L%16, group g = L/16; halves e<8 -> K=k0+8g+e, e>=8 -> K=k0+16+8g+(e-8).
static __device__ __forceinline__ v16h a_frag_lds(const _Float16* hb, int m, int k0, int g) {
  const int off = m * HSTR + k0 + 8 * g;
  v8h lo = *(const v8h*)(hb + off);
  v8h hi = *(const v8h*)(hb + off + 16);
  return cat16(lo, hi);
}

// A fragment from LDS f32 matrix (staged x_t), converted to f16 in registers.
static __device__ __forceinline__ v16h a_frag_lds_f32(const float* xb, int m, int k0, int g) {
  const int off = m * XSTR + k0 + 8 * g;
  f4 u0 = *(const f4*)(xb + off);
  f4 u1 = *(const f4*)(xb + off + 4);
  f4 u2 = *(const f4*)(xb + off + 16);
  f4 u3 = *(const f4*)(xb + off + 20);
  h4 a0 = __builtin_convertvector(u0, h4);
  h4 a1 = __builtin_convertvector(u1, h4);
  h4 a2 = __builtin_convertvector(u2, h4);
  h4 a3 = __builtin_convertvector(u3, h4);
  v8h lo = __builtin_shufflevector(a0, a1, 0,1,2,3,4,5,6,7);
  v8h hi = __builtin_shufflevector(a2, a3, 0,1,2,3,4,5,6,7);
  return cat16(lo, hi);
}

// B fragment (32x16 f16): column n of W^T == row n of row-major W[NG][H].
// Lane n = n0 + L%16, g = L/16, halves e<8 -> K = k0 + 8g + e, e>=8 -> K = k0 + 16 + 8g + (e-8): two 16B global loads.
static __device__ __forceinline__ v16h b_frag(GAS const _Float16* w, int n, int k0, int g) {
  GAS const _Float16* p = w + (size_t)n * H + k0 + 8 * g;
  return cat16(*(GAS const v8h*)p, *(GAS const v8h*)(p + 16));
}

static __device__ __forceinline__ float sigmoidf(float v) {
  return 1.0f / (1.0f + __expf(-v));
}

// ---------------------------------------------------------------------------
// Prologue: convert the 6 weight matrices fp32 -> f16 into workspace.
// Layout in ws: [Wih0, Whh0, Wih1, Whh1, Wih2, Whh2], each NG x H row-major.
// ---------------------------------------------------------------------------
__global__ void cvt_weights(const float* __restrict__ a, const float* __restrict__ b,
                            const float* __restrict__ c, const float* __restrict__ d,
                            const float* __restrict__ e, const float* __restrict__ f,
                            _Float16* __restrict__ dst) {
  int i = blockIdx.x * 256 + threadIdx.x;
  if (i >= 6 * WPER) return;
  int which = i / WPER;
  int off   = i % WPER;
  const float* s = (which == 0) ? a : (which == 1) ? b : (which == 2) ? c
                 : (which == 3) ? d : (which == 4) ? e : f;
  dst[i] = (_Float16)s[off];
}

// ---------------------------------------------------------------------------
// Persistent GRU: one workgroup per 16-row batch tile. Hidden state, gate
// exchange, and the staged x_t tile all live in LDS; weights stream from L2.
// x_{t+1} is prefetched into the other LDS buffer via async global->LDS.
// ---------------------------------------------------------------------------
__global__ void __launch_bounds__(1024, 1)
gru3_persistent(const float* __restrict__ x_, const _Float16* __restrict__ wf16_,
                const float* __restrict__ bih0, const float* __restrict__ bhh0,
                const float* __restrict__ bih1, const float* __restrict__ bhh1,
                const float* __restrict__ bih2, const float* __restrict__ bhh2,
                float* __restrict__ out_) {
  __shared__ float    s_gsum[16 * GSTR];      // gi+gh for r (cols 0..511) and z (512..1023)
  __shared__ _Float16 s_in[16 * NSTR];        // i_n
  __shared__ _Float16 s_hn[16 * NSTR];        // h_n
  __shared__ _Float16 s_h[3][16 * HSTR];      // hidden states h0,h1,h2 (f16)
  __shared__ float    s_x[2][16 * XSTR];      // double-buffered x_t tile (f32)

  const int tid  = threadIdx.x;
  const int lane = tid & 31;
  const int wv   = tid >> 5;        // wave 0..31
  const int mf   = lane & 15;       // A row within tile / C column
  const int gf   = lane >> 4;       // K group / C row group
  const int m0   = blockIdx.x * 16; // batch-tile base row

  // Explicit global address space: force global_load/global_store (not FLAT).
  GAS const float*    xg   = (GAS const float*)x_;
  GAS float*          outg = (GAS float*)out_;
  GAS const _Float16* wg   = (GAS const _Float16*)wf16_;
  GAS const _Float16* wih[3] = { wg,            wg + 2 * WPER, wg + 4 * WPER };
  GAS const _Float16* whh[3] = { wg + WPER,     wg + 3 * WPER, wg + 5 * WPER };
  GAS const float*    bih[3] = { (GAS const float*)bih0, (GAS const float*)bih1, (GAS const float*)bih2 };
  GAS const float*    bhh[3] = { (GAS const float*)bhh0, (GAS const float*)bhh1, (GAS const float*)bhh2 };

  // zero-init hidden states
  _Float16* hflat = &s_h[0][0];
  for (int i = tid; i < 3 * 16 * HSTR; i += 1024) hflat[i] = (_Float16)0.0f;

  // Stage x_tt tile (16 rows x 512 f32) into s_x[buf]: 2048 x 16B chunks.
  auto stage_x = [&](int buf, int tt) {
    for (int c = tid; c < 2048; c += 1024) {
      int r   = c >> 7;              // batch row 0..15
      int col = (c & 127) << 2;      // float column, multiple of 4
      size_t goff = ((size_t)(m0 + r) * T + tt) * H + col;
      float* lp = &s_x[buf][r * XSTR + col];
#if HAVE_ASYNC_LDS
      __builtin_amdgcn_global_load_async_to_lds_b128(
          (GAS i4v*)(xg + goff), (LAS i4v*)lp, 0, 0);
#else
      *(f4*)lp = *(GAS const f4*)(xg + goff);
#endif
    }
  };
  auto wait_x = [&]() {
#if HAVE_ASYNC_LDS
    __builtin_amdgcn_s_wait_asynccnt(0);
#endif
  };

  stage_x(0, 0);                    // prime the pipeline with x_0

  // update-phase mapping: thread -> (row um, cols uc + 64*e): bank-conflict-free
  const int um = tid >> 6;   // 0..15
  const int uc = tid & 63;   // 0..63

  for (int t = 0; t < T; ++t) {
    const int xb = t & 1;
    wait_x();                       // x_t landed in s_x[xb]
    __syncthreads();                // ... and is visible to all waves
    if (t + 1 < T) stage_x(xb ^ 1, t + 1);   // async prefetch overlaps all of step t

#pragma unroll
    for (int l = 0; l < 3; ++l) {
      // ---- dual GEMM: gi = in_l @ Wih_l^T + bih_l ; gh = h_l @ Whh_l^T + bhh_l ----
      v8f ai[3], ah[3];
#pragma unroll
      for (int j = 0; j < 3; ++j) {
        int n = (wv * 3 + j) * 16 + mf;
        v8f zz = {};
        ai[j] = zz + bih[l][n];     // bias depends only on column -> splat
        ah[j] = zz + bhh[l][n];
      }
      for (int kt = 0; kt < 16; ++kt) {
        const int k0 = kt * 32;
        v16h a_i = (l == 0) ? a_frag_lds_f32(s_x[xb], mf, k0, gf)
                            : a_frag_lds(s_h[l - 1], mf, k0, gf);
        v16h a_h = a_frag_lds(s_h[l], mf, k0, gf);
#pragma unroll
        for (int j = 0; j < 3; ++j) {
          int n = (wv * 3 + j) * 16 + mf;
          v16h b_i = b_frag(wih[l], n, k0, gf);
          v16h b_h = b_frag(whh[l], n, k0, gf);
          ai[j] = wmma16x16x32(a_i, b_i, ai[j]);
          ah[j] = wmma16x16x32(a_h, b_h, ah[j]);
        }
      }
      // ---- write gate pieces to LDS (wave-uniform branch: nt<64 <=> r/z region) ----
#pragma unroll
      for (int j = 0; j < 3; ++j) {
        int nt = wv * 3 + j;
        int n  = nt * 16 + mf;
        if (nt < 64) {
#pragma unroll
          for (int i = 0; i < 8; ++i) {
            int m = i + 8 * gf;                    // C layout: VGPR i -> row i + 8*(lane/16)
            s_gsum[m * GSTR + n] = ai[j][i] + ah[j][i];
          }
        } else {
          int c = n - 1024;
#pragma unroll
          for (int i = 0; i < 8; ++i) {
            int m = i + 8 * gf;
            s_in[m * NSTR + c] = (_Float16)ai[j][i];
            s_hn[m * NSTR + c] = (_Float16)ah[j][i];
          }
        }
      }
      __syncthreads();

      // ---- gate update: 8192 elements / 1024 threads ----
#pragma unroll
      for (int e = 0; e < 8; ++e) {
        int jj = uc + 64 * e;
        float r  = sigmoidf(s_gsum[um * GSTR + jj]);
        float z  = sigmoidf(s_gsum[um * GSTR + 512 + jj]);
        float nn = tanhf((float)s_in[um * NSTR + jj] + r * (float)s_hn[um * NSTR + jj]);
        float ho = (float)s_h[l][um * HSTR + jj];
        float hv = (1.0f - z) * nn + z * ho;
        if (l == 2) {
          float xv = s_x[xb][um * XSTR + jj];      // mask from the staged exact f32 x_t
          hv = (xv != 0.0f) ? hv : 0.0f;
          outg[((size_t)(m0 + um) * T + t) * H + jj] = hv;
        }
        s_h[l][um * HSTR + jj] = (_Float16)hv;
      }
      __syncthreads();
    }
  }
}

extern "C" void kernel_launch(void* const* d_in, const int* in_sizes, int n_in,
                              void* d_out, int out_size, void* d_ws, size_t ws_size,
                              hipStream_t stream) {
  (void)in_sizes; (void)n_in; (void)out_size; (void)ws_size;
  const float* x = (const float*)d_in[0];
  _Float16* wf16 = (_Float16*)d_ws;   // 6 * 786432 f16 = 9.4 MB

  // Convert weights fp32 -> f16 (order: Wih0, Whh0, Wih1, Whh1, Wih2, Whh2)
  int ntot = 6 * WPER;
  cvt_weights<<<(ntot + 255) / 256, 256, 0, stream>>>(
      (const float*)d_in[1], (const float*)d_in[2],
      (const float*)d_in[5], (const float*)d_in[6],
      (const float*)d_in[9], (const float*)d_in[10], wf16);

  // Persistent scan: 8 batch tiles, 1024 threads (32 waves) each
  gru3_persistent<<<dim3(B / 16), dim3(1024), 0, stream>>>(
      x, wf16,
      (const float*)d_in[3],  (const float*)d_in[4],
      (const float*)d_in[7],  (const float*)d_in[8],
      (const float*)d_in[11], (const float*)d_in[12],
      (float*)d_out);
}
